// QKMProjLayer_2568390443653
// MI455X (gfx1250) — matrix operand order repaired and checked by reading, not compiled, not run on hardware
//
#include <hip/hip_runtime.h>
#include <hip/hip_bf16.h>

// ---------------------------------------------------------------------------
// QKMProjLayer on MI455X (gfx1250, wave32, WMMA + async LDS DMA)
// out[b] = sum_j comp_w[j] * exp(-2*d2[b,j]),  d2 = ||x||^2 + ||c||^2 - 2 x.c
// GEMM in bf16x3 (hi/lo split) with v_wmma_f32_16x16x32_bf16; tiles staged
// with global_load_async_to_lds_b128 (ASYNCcnt) into double-buffered LDS.
// ---------------------------------------------------------------------------

typedef __attribute__((ext_vector_type(16))) __bf16 bf16x16;
typedef __attribute__((ext_vector_type(4)))  __bf16 bf16x4;
typedef __attribute__((ext_vector_type(8)))  float  floatx8;
typedef __attribute__((ext_vector_type(4)))  unsigned int uintx4;

union Frag16 { bf16x16 v; uintx4 u[2]; };

#define BM 128
#define BN 128
#define BK 32
#define LDA 40   // padded LDS row stride (elems): 80B -> conflict-free b128 rows

// one async b128 pair: copies 32B, INST_OFFSET applies to both LDS and global
#define ASYNC_CP32(ldsaddr, gaddr)                                             \
    asm volatile("global_load_async_to_lds_b128 %0, %1, off"                   \
                 :: "v"(ldsaddr), "v"(gaddr) : "memory");                      \
    asm volatile("global_load_async_to_lds_b128 %0, %1, off offset:16"         \
                 :: "v"(ldsaddr), "v"(gaddr) : "memory");

// ---------------------------------------------------------------------------
// Kernel 1: row sums-of-squares + fp32 -> (bf16 hi, bf16 lo) split
// ---------------------------------------------------------------------------
__global__ void qkm_prep_rows(const float* __restrict__ x,
                              const float* __restrict__ cx,
                              __bf16* __restrict__ xhi, __bf16* __restrict__ xlo,
                              __bf16* __restrict__ chi, __bf16* __restrict__ clo,
                              float* __restrict__ x2, float* __restrict__ c2,
                              int B, int D) {
    const int tid  = threadIdx.x;
    const int lane = tid & 31;
    const int wid  = tid >> 5;
    int row = blockIdx.x;
    const float* src; __bf16* hi; __bf16* lo; float* nrm;
    if (row < B) { src = x;  hi = xhi; lo = xlo; nrm = x2; }
    else { row -= B; src = cx; hi = chi; lo = clo; nrm = c2; }

    float s = 0.0f;
    for (int c = tid * 4; c < D; c += 256 * 4) {
        const size_t off = (size_t)row * D + c;
        float4 v = *(const float4*)&src[off];
        s += v.x * v.x + v.y * v.y + v.z * v.z + v.w * v.w;
        bf16x4 hv, lv;
        hv.x = (__bf16)v.x; lv.x = (__bf16)(v.x - (float)hv.x);
        hv.y = (__bf16)v.y; lv.y = (__bf16)(v.y - (float)hv.y);
        hv.z = (__bf16)v.z; lv.z = (__bf16)(v.z - (float)hv.z);
        hv.w = (__bf16)v.w; lv.w = (__bf16)(v.w - (float)hv.w);
        *(bf16x4*)&hi[off] = hv;
        *(bf16x4*)&lo[off] = lv;
    }
    for (int off = 16; off; off >>= 1) s += __shfl_xor(s, off, 32);
    __shared__ float red[8];
    if (lane == 0) red[wid] = s;
    __syncthreads();
    if (tid == 0) {
        float t = 0.0f;
        for (int j = 0; j < 8; ++j) t += red[j];
        nrm[row] = t;
    }
}

// ---------------------------------------------------------------------------
// Kernel 2: wsum = sum_j (|c_w[j]| + eps)   (single block)
// ---------------------------------------------------------------------------
__global__ void qkm_reduce_w(const float* __restrict__ cw, float* __restrict__ wsum, int n) {
    const int tid = threadIdx.x, lane = tid & 31, wid = tid >> 5;
    float s = 0.0f;
    for (int i = tid; i < n; i += 256) s += fabsf(cw[i]) + 1e-10f;
    for (int off = 16; off; off >>= 1) s += __shfl_xor(s, off, 32);
    __shared__ float red[8];
    if (lane == 0) red[wid] = s;
    __syncthreads();
    if (tid == 0) {
        float t = 0.0f;
        for (int j = 0; j < 8; ++j) t += red[j];
        wsum[0] = t;
    }
}

// ---------------------------------------------------------------------------
// Kernel 3: bf16x3 WMMA GEMM, async double-buffered staging, RBF epilogue.
// Grid: (N/BN, B/BM), 256 threads = 8 waves (4 m x 2 n), 24 WMMA / K-step.
// ---------------------------------------------------------------------------
__global__ void __launch_bounds__(256)
qkm_gemm_rbf(const __bf16* __restrict__ xhi, const __bf16* __restrict__ xlo,
             const __bf16* __restrict__ chi, const __bf16* __restrict__ clo,
             const float* __restrict__ x2g, const float* __restrict__ c2g,
             const float* __restrict__ cw,  float* __restrict__ accum,
             int D) {
    // [stage][hi/lo][tile]
    __shared__ __align__(16) __bf16 Asm[2][2][BM * LDA];
    __shared__ __align__(16) __bf16 Bsm[2][2][BN * LDA];

    const int rowBase = blockIdx.y * BM;
    const int colBase = blockIdx.x * BN;

    const int tid    = threadIdx.x;
    const int lane   = tid & 31;
    const int lane16 = lane & 15;
    const int half   = lane >> 4;
    const int wid    = tid >> 5;
    const int wm0    = (wid & 3) * 32;   // wave m-origin (4 waves)
    const int wn0    = (wid >> 2) * 64;  // wave n-origin (2 waves)

    // staging: 2 threads per row, one 32B chunk each (8 async b128 / stage)
    const int srow = tid >> 1;
    const int sseg = (tid & 1) * 16;
    const int lofs = srow * LDA + sseg;

    auto issue_stage = [&](int stage, int buf) {
        const size_t gx = (size_t)(rowBase + srow) * D + stage * BK + sseg;
        const size_t gc = (size_t)(colBase + srow) * D + stage * BK + sseg;
        unsigned la;
        unsigned long long ga;
        la = (unsigned)(uintptr_t)&Asm[buf][0][lofs];
        ga = (unsigned long long)(uintptr_t)(xhi + gx);
        ASYNC_CP32(la, ga);
        la = (unsigned)(uintptr_t)&Asm[buf][1][lofs];
        ga = (unsigned long long)(uintptr_t)(xlo + gx);
        ASYNC_CP32(la, ga);
        la = (unsigned)(uintptr_t)&Bsm[buf][0][lofs];
        ga = (unsigned long long)(uintptr_t)(chi + gc);
        ASYNC_CP32(la, ga);
        la = (unsigned)(uintptr_t)&Bsm[buf][1][lofs];
        ga = (unsigned long long)(uintptr_t)(clo + gc);
        ASYNC_CP32(la, ga);
    };

    floatx8 acc[2][4] = {};
    const int ksteps = D / BK;

    issue_stage(0, 0);
    if (ksteps > 1) issue_stage(1, 1);

    for (int kt = 0; kt < ksteps; ++kt) {
        const int cur = kt & 1;
        // stage kt is the oldest in flight; asyncs retire in order
        if (kt + 1 < ksteps) {
            asm volatile("s_wait_asynccnt 0x8" ::: "memory");
        } else {
            asm volatile("s_wait_asynccnt 0x0" ::: "memory");
        }
        __syncthreads();  // everyone's stage-kt data landed in LDS

        // A fragments: 16-bit A 16x32 layout -> two 16B chunks per lane
        const __bf16* Ah = Asm[cur][0];
        const __bf16* Al = Asm[cur][1];
        Frag16 ahi[2], alo[2];
#pragma unroll
        for (int mi = 0; mi < 2; ++mi) {
            const int r = (wm0 + mi * 16 + lane16) * LDA;
            ahi[mi].u[0] = *(const uintx4*)&Ah[r + half * 8];
            ahi[mi].u[1] = *(const uintx4*)&Ah[r + 16 + half * 8];
            alo[mi].u[0] = *(const uintx4*)&Al[r + half * 8];
            alo[mi].u[1] = *(const uintx4*)&Al[r + 16 + half * 8];
        }
        // B fragments: 16-bit B 32x16 layout -> two 16B chunks per lane
        const __bf16* Bh = Bsm[cur][0];
        const __bf16* Bl = Bsm[cur][1];
        Frag16 bhi[4], blo[4];
#pragma unroll
        for (int ni = 0; ni < 4; ++ni) {
            const int r = (wn0 + ni * 16 + lane16) * LDA + half * 8;
            bhi[ni].u[0] = *(const uintx4*)&Bh[r];
            bhi[ni].u[1] = *(const uintx4*)&Bh[r + 16];
            blo[ni].u[0] = *(const uintx4*)&Bl[r];
            blo[ni].u[1] = *(const uintx4*)&Bl[r + 16];
        }

#pragma unroll
        for (int mi = 0; mi < 2; ++mi)
#pragma unroll
            for (int ni = 0; ni < 4; ++ni) {
                acc[mi][ni] = __builtin_amdgcn_wmma_f32_16x16x32_bf16(
                    false, ahi[mi].v, false, bhi[ni].v, (short)0, acc[mi][ni], false, false);
                acc[mi][ni] = __builtin_amdgcn_wmma_f32_16x16x32_bf16(
                    false, ahi[mi].v, false, blo[ni].v, (short)0, acc[mi][ni], false, false);
                acc[mi][ni] = __builtin_amdgcn_wmma_f32_16x16x32_bf16(
                    false, alo[mi].v, false, bhi[ni].v, (short)0, acc[mi][ni], false, false);
            }

        __syncthreads();  // all waves done reading buf[cur]
        if (kt + 2 < ksteps) issue_stage(kt + 2, cur);  // refill freed buffer
    }

    // Epilogue: d2 = x2 + c2 - 2*xc ; contribution w_raw * exp(-2*d2)
    float c2v[4], wv[4];
#pragma unroll
    for (int ni = 0; ni < 4; ++ni) {
        const int gn = colBase + wn0 + ni * 16 + lane16;
        c2v[ni] = c2g[gn];
        wv[ni]  = fabsf(cw[gn]) + 1e-10f;
    }
#pragma unroll
    for (int mi = 0; mi < 2; ++mi) {
#pragma unroll
        for (int i = 0; i < 8; ++i) {
            const int gm = rowBase + wm0 + mi * 16 + half * 8 + i;
            const float xx = x2g[gm];
            float s = 0.0f;
#pragma unroll
            for (int ni = 0; ni < 4; ++ni) {
                const float xc = acc[mi][ni][i];
                const float d2 = xx + c2v[ni] - 2.0f * xc;
                s += wv[ni] * __expf(-2.0f * d2);   // k^2 = exp(-2*gamma*d2)
            }
            for (int off = 8; off; off >>= 1) s += __shfl_xor(s, off, 16);
            if (lane16 == 0) atomicAdd(&accum[gm], s);
        }
    }
}

// ---------------------------------------------------------------------------
// Kernel 4: out[b] = accum[b] / wsum
// ---------------------------------------------------------------------------
__global__ void qkm_finalize(const float* __restrict__ accum,
                             const float* __restrict__ wsum,
                             float* __restrict__ out, int B) {
    const int i = blockIdx.x * blockDim.x + threadIdx.x;
    if (i < B) out[i] = accum[i] / wsum[0];
}

// ---------------------------------------------------------------------------
extern "C" void kernel_launch(void* const* d_in, const int* in_sizes, int n_in,
                              void* d_out, int out_size, void* d_ws, size_t ws_size,
                              hipStream_t stream) {
    (void)n_in; (void)out_size; (void)ws_size;
    const float* x  = (const float*)d_in[0];
    const float* cx = (const float*)d_in[1];
    const float* cw = (const float*)d_in[2];
    float* out = (float*)d_out;

    const int D = 1024;
    const int B = in_sizes[0] / D;
    const int N = in_sizes[2];

    char* ws = (char*)d_ws;
    const size_t szX = (size_t)B * D * sizeof(__bf16);
    const size_t szC = (size_t)N * D * sizeof(__bf16);
    __bf16* xhi = (__bf16*)(ws);                     size_t off = szX;
    __bf16* xlo = (__bf16*)(ws + off);               off += szX;
    __bf16* chi = (__bf16*)(ws + off);               off += szC;
    __bf16* clo = (__bf16*)(ws + off);               off += szC;
    float*  x2  = (float*)(ws + off);                off += (size_t)B * sizeof(float);
    float*  c2  = (float*)(ws + off);                off += (size_t)N * sizeof(float);
    float*  acc = (float*)(ws + off);                off += (size_t)B * sizeof(float);
    float*  wsum = (float*)(ws + off);

    hipMemsetAsync(acc, 0, (size_t)B * sizeof(float), stream);

    qkm_prep_rows<<<B + N, 256, 0, stream>>>(x, cx, xhi, xlo, chi, clo, x2, c2, B, D);
    qkm_reduce_w<<<1, 256, 0, stream>>>(cw, wsum, N);

    dim3 grid(N / BN, B / BM);
    qkm_gemm_rbf<<<grid, 256, 0, stream>>>(xhi, xlo, chi, clo, x2, c2, cw, acc, D);

    qkm_finalize<<<(B + 255) / 256, 256, 0, stream>>>(acc, wsum, out, B);
}
